// MultiHeadAttention_18056042512907
// MI455X (gfx1250) — compile-verified, matrix operand order checked by reading, not run on hardware
//
#include <hip/hip_runtime.h>

// ---------------------------------------------------------------------------
// MHA forward for MI455X (gfx1250): f32 in -> bf16 WMMA (f32 accum) -> f32 out
//   B=4, S(len_q=len_k)=2048, D_MODEL=1024, H=16, Dk=Dv=64
// Round 4: all LDS fragment gathers forced to ds_load_b128 via explicit
//          aligned uint4 loads; P bounce buffer moved to swizzled layout.
// ---------------------------------------------------------------------------

#define BATCH   4
#define SEQ     2048
#define DM      1024
#define NH      16
#define DH      64
#define NEGV    (-1000000000.0f)

typedef __attribute__((ext_vector_type(16))) __bf16       v16bf;
typedef __attribute__((ext_vector_type(8)))  float        v8f;
typedef __attribute__((ext_vector_type(8)))  unsigned int v8u;

// round-to-nearest-even f32 -> bf16 (raw ushort)
static __device__ __forceinline__ unsigned short f2bf(float f) {
    unsigned int u = __builtin_bit_cast(unsigned int, f);
    u += 0x7FFFu + ((u >> 16) & 1u);
    return (unsigned short)(u >> 16);
}

// LDS byte offset of a generic pointer into __shared__ memory: flat LDS
// addresses carry the LDS offset in addr[31:0] (ISA 10.2 aperture mapping).
static __device__ __forceinline__ unsigned lds32(const void* p) {
    return (unsigned)(unsigned long long)p;
}

// GLOBAL_LOAD_ASYNC_TO_LDS_B128, GVS mode: mem = SGPR64 + VGPR32.
// vdst VGPR supplies the LDS byte address. Tracked by ASYNCcnt.
static __device__ __forceinline__ void async_ld_b128(unsigned lds_off,
                                                     unsigned voff_bytes,
                                                     const void* sbase) {
    asm volatile("global_load_async_to_lds_b128 %0, %1, %2"
                 :
                 : "v"(lds_off), "v"(voff_bytes), "s"(sbase)
                 : "memory");
}

static __device__ __forceinline__ void wait_async0() {
    asm volatile("s_wait_asynccnt 0" ::: "memory");
}

// XOR chunk swizzles (chunk = 16B = 4 u32). Bank-conflict-free for 16-lane
// row gathers: banks = rowbase + 4*(c^swz) + w enumerate 16 distinct values.
static __device__ __forceinline__ int swz4(int row) {        // 4-chunk rows
    return (row & 3) ^ ((row >> 2) & 3);
}
static __device__ __forceinline__ int swz8(int row) {        // 8-chunk rows
    return (row >> 1) & 7;
}

static __device__ __forceinline__ v8u pack2(uint4 a, uint4 b) {
    v8u r;
    r[0] = a.x; r[1] = a.y; r[2] = a.z; r[3] = a.w;
    r[4] = b.x; r[5] = b.y; r[6] = b.z; r[7] = b.w;
    return r;
}

// CDNA5 16-bit A/B fragment gather (16x32 tile, K-pairs per u32):
//   lane L: group g = L>>4; VGPR j<4 hold K = 8g+2j .. ; j>=4: K = 16+8g+..
// Each half-fragment is one 16B-aligned chunk -> explicit uint4 (ds_load_b128).
static __device__ __forceinline__ v16bf ld_frag_sw4(const unsigned int* base,
                                                    int row, int lane) {
    const int g = (lane >> 4) & 1, x = swz4(row);
    const unsigned int* rp = base + row * 16;
    const uint4 a = *(const uint4*)(rp + (((g ^ x) & 3) << 2));
    const uint4 b = *(const uint4*)(rp + ((((2 + g) ^ x) & 3) << 2));
    return __builtin_bit_cast(v16bf, pack2(a, b));
}
// colc: chunk offset of the K-dim half (0 for d[0:32), 4 for d[32:64))
static __device__ __forceinline__ v16bf ld_frag_sw8(const unsigned int* base,
                                                    int row, int colc, int lane) {
    const int g = (lane >> 4) & 1, x = swz8(row);
    const unsigned int* rp = base + row * 32;
    const uint4 a = *(const uint4*)(rp + ((((colc + g) ^ x) & 7) << 2));
    const uint4 b = *(const uint4*)(rp + ((((colc + 2 + g) ^ x) & 7) << 2));
    return __builtin_bit_cast(v16bf, pack2(a, b));
}

static __device__ __forceinline__ v8f wmma_bf16(v16bf a, v16bf b, v8f c) {
    return __builtin_amdgcn_wmma_f32_16x16x32_bf16(
        false, a, false, b, (short)0, c, false, false);
}

// ---------------------------------------------------------------------------
// Kernel 1: f32 -> bf16 conversion (n divisible by 1024)
// ---------------------------------------------------------------------------
__global__ __launch_bounds__(256) void mha_cvt_bf16(const float* __restrict__ in,
                                                    unsigned short* __restrict__ out) {
    const int i = (blockIdx.x * 256 + threadIdx.x) * 4;
    const float4 v = *(const float4*)(in + i);
    ushort4 o;
    o.x = f2bf(v.x); o.y = f2bf(v.y); o.z = f2bf(v.z); o.w = f2bf(v.w);
    *(ushort4*)(out + i) = o;
}

// ---------------------------------------------------------------------------
// Kernel 2: projection GEMM  C[8192 x 1024] = A[8192 x 1024] * W^T (+bias)
//   Block: 256 thr (8 waves), tile 256(M) x 64(N=one head), K-step 32.
//   Each wave: 32x64 strip = 2 A-frags x 4 B-frags = 8 WMMAs / K-step.
//   Double-buffered swizzled LDS, b128 async staging overlapped with compute.
//   Output bf16: transposed==0 -> [B][H][S][64] ; transposed==1 -> [B][H][64][S]
// ---------------------------------------------------------------------------
__global__ __launch_bounds__(256) void mha_proj_gemm(
    const unsigned short* __restrict__ A, const unsigned short* __restrict__ W,
    const float* __restrict__ bias, unsigned short* __restrict__ out,
    int transposed) {
    __shared__ __align__(16) unsigned int sA[2][256 * 16];  // 256 x 32 bf16, swz4
    __shared__ __align__(16) unsigned int sB[2][64 * 16];   // 64  x 32 bf16, swz4

    const int tid  = threadIdx.x;
    const int wave = tid >> 5;
    const int lane = tid & 31;
    const int n    = lane & 15;
    const int g    = lane >> 4;
    const int mBase = blockIdx.x * 256;
    const int head  = blockIdx.y;

    const unsigned int* Au = (const unsigned int*)A;   // row stride 512 u32
    const unsigned int* Wu = (const unsigned int*)W;

    v8f accA[4] = {}, accB[4] = {};

    // --- b128 async staging of one K-step tile pair into buffer `buf` ---
    auto stage = [&](int buf, int kk) {
        const int ku = kk >> 1;                 // multiple of 16 -> 16B aligned
#pragma unroll
        for (int p = 0; p < 4; ++p) {           // A: 256 rows x 4 chunks
            const int idx = p * 256 + tid;
            const int row = idx >> 2, c = idx & 3;
            async_ld_b128(lds32(&sA[buf][row * 16 + ((c ^ swz4(row)) << 2)]),
                          (unsigned)(((mBase + row) * 512 + ku + c * 4) * 4), Au);
        }
        {                                       // W: 64 rows x 4 chunks
            const int row = tid >> 2, c = tid & 3;
            async_ld_b128(lds32(&sB[buf][row * 16 + ((c ^ swz4(row)) << 2)]),
                          (unsigned)(((head * 64 + row) * 512 + ku + c * 4) * 4), Wu);
        }
    };

    stage(0, 0);
    wait_async0();
    __syncthreads();

    for (int kk = 0; kk < DM; kk += 32) {
        const int buf = (kk >> 5) & 1;
        if (kk + 32 < DM) stage(buf ^ 1, kk + 32);   // prefetch next tile

        const v16bf af0 = ld_frag_sw4(&sA[buf][0], wave * 32 + n,      lane);
        const v16bf af1 = ld_frag_sw4(&sA[buf][0], wave * 32 + 16 + n, lane);
#pragma unroll
        for (int s4 = 0; s4 < 4; ++s4) {
            const v16bf bf = ld_frag_sw4(&sB[buf][0], s4 * 16 + n, lane);
            accA[s4] = wmma_bf16(af0, bf, accA[s4]);
            accB[s4] = wmma_bf16(af1, bf, accB[s4]);
        }
        wait_async0();
        __syncthreads();
    }

#pragma unroll
    for (int s4 = 0; s4 < 4; ++s4) {
        const int d = s4 * 16 + n;
        const float bv = bias ? bias[head * 64 + d] : 0.0f;
#pragma unroll
        for (int half = 0; half < 2; ++half) {
            const v8f* acc = half ? accB : accA;
#pragma unroll
            for (int r = 0; r < 8; ++r) {
                const int mrow = mBase + wave * 32 + half * 16 + r + 8 * g;
                const int b = mrow >> 11, sloc = mrow & (SEQ - 1);
                const unsigned short h = f2bf(acc[s4][r] + bv);
                if (!transposed)
                    out[(((b * NH + head) * SEQ) + sloc) * DH + d] = h;
                else
                    out[((b * NH + head) * DH + d) * SEQ + sloc] = h;
            }
        }
    }
}

// ---------------------------------------------------------------------------
// Kernel 3: flash attention.
//   grid (SEQ/64, B*H), block 128 (4 waves, 16 queries each).
//   Double-buffered swizzled K/V tiles staged by b128 async-to-LDS.
//   qp/kp: bf16 [B][H][S][64]; vt: bf16 [B][H][64][S]; out: f32 [B][S][H*64]
// ---------------------------------------------------------------------------
__global__ __launch_bounds__(128) void mha_attn(
    const unsigned short* __restrict__ qp, const unsigned short* __restrict__ kp,
    const unsigned short* __restrict__ vt, const int* __restrict__ mask,
    float* __restrict__ out) {
    __shared__ __align__(16) unsigned int sK[2][32 * 32];  // 32 keys x 64 d, swz8
    __shared__ __align__(16) unsigned int sV[2][64 * 16];  // 64 d x 32 keys, swz4
    __shared__ __align__(16) unsigned int sP[4 * 16 * 16]; // per-wave P, swz4

    const int tid  = threadIdx.x;
    const int wave = tid >> 5;
    const int lane = tid & 31;
    const int g    = lane >> 4;
    const int n    = lane & 15;

    const int bh = blockIdx.y;
    const int b  = bh >> 4;
    const int h  = bh & 15;
    const int qrow = blockIdx.x * 64 + wave * 16 + n;

    const unsigned int* qu = (const unsigned int*)qp;  // (bh*S+s)*32 + d/2
    const unsigned int* ku = (const unsigned int*)kp;
    const unsigned int* vu = (const unsigned int*)vt;  // (bh*64+d)*1024 + s/2

    // q A-fragments for K-dim halves [0,32) and [32,64)
    v8u q0r, q1r;
#pragma unroll
    for (int j = 0; j < 8; ++j) {
        const int k0 = (j < 4) ? (8 * g + 2 * j) : (16 + 8 * g + 2 * (j - 4));
        const int rowoff = (bh * SEQ + qrow) * 32;
        q0r[j] = qu[rowoff + (k0 >> 1)];
        q1r[j] = qu[rowoff + ((k0 + 32) >> 1)];
    }
    const v16bf Aq0 = __builtin_bit_cast(v16bf, q0r);
    const v16bf Aq1 = __builtin_bit_cast(v16bf, q1r);

    v8f acc[4] = {};
    float mrow[8], lrow[8];
#pragma unroll
    for (int r = 0; r < 8; ++r) { mrow[r] = -3.0e38f; lrow[r] = 0.0f; }

    auto stage = [&](int buf, int kb) {
#pragma unroll
        for (int p = 0; p < 2; ++p) {           // K tile: 32 rows x 8 chunks
            const int idx = p * 128 + tid;
            const int row = idx >> 3, c = idx & 7;
            async_ld_b128(lds32(&sK[buf][row * 32 + ((c ^ swz8(row)) << 2)]),
                          (unsigned)(((bh * SEQ + kb + row) * 32 + c * 4) * 4), ku);
        }
#pragma unroll
        for (int p = 0; p < 2; ++p) {           // V tile: 64 rows x 4 chunks
            const int idx = p * 128 + tid;
            const int row = idx >> 2, c = idx & 3;
            async_ld_b128(lds32(&sV[buf][row * 16 + ((c ^ swz4(row)) << 2)]),
                          (unsigned)(((bh * DH + row) * 1024 + (kb >> 1) + c * 4) * 4), vu);
        }
    };

    stage(0, 0);
    wait_async0();
    __syncthreads();

    for (int kb = 0; kb < SEQ; kb += 32) {
        const int buf = (kb >> 5) & 1;
        if (kb + 32 < SEQ) stage(buf ^ 1, kb + 32);   // prefetch next tiles

        // S = Q * K^T  (two 16x16 key sub-tiles, each 2 WMMAs over d)
        v8f S0 = {}, S1 = {};
        S0 = wmma_bf16(Aq0, ld_frag_sw8(&sK[buf][0], n,      0, lane), S0);
        S0 = wmma_bf16(Aq1, ld_frag_sw8(&sK[buf][0], n,      4, lane), S0);
        S1 = wmma_bf16(Aq0, ld_frag_sw8(&sK[buf][0], n + 16, 0, lane), S1);
        S1 = wmma_bf16(Aq1, ld_frag_sw8(&sK[buf][0], n + 16, 4, lane), S1);

        // scale + key mask (mask is per-column == per-lane)
        const int mk0 = mask[b * SEQ + kb + n];
        const int mk1 = mask[b * SEQ + kb + 16 + n];
        float P0[8], P1[8], t[8];
#pragma unroll
        for (int r = 0; r < 8; ++r) {
            float s0 = S0[r] * 0.125f;          // 1/sqrt(64)
            float s1 = S1[r] * 0.125f;
            if (!mk0) s0 = NEGV;
            if (!mk1) s1 = NEGV;
            S0[r] = s0; S1[r] = s1;
            t[r] = fmaxf(s0, s1);
        }
#pragma unroll
        for (int r = 0; r < 8; ++r) {           // row-max across 16 lanes
            t[r] = fmaxf(t[r], __shfl_xor(t[r], 1, 32));
            t[r] = fmaxf(t[r], __shfl_xor(t[r], 2, 32));
            t[r] = fmaxf(t[r], __shfl_xor(t[r], 4, 32));
            t[r] = fmaxf(t[r], __shfl_xor(t[r], 8, 32));
        }
        float rs[8];
#pragma unroll
        for (int r = 0; r < 8; ++r) {           // online softmax update
            const float mn = fmaxf(mrow[r], t[r]);
            const float al = __expf(mrow[r] - mn);
            mrow[r] = mn;
            lrow[r] *= al;
            P0[r] = __expf(S0[r] - mn);
            P1[r] = __expf(S1[r] - mn);
            rs[r] = P0[r] + P1[r];
#pragma unroll
            for (int s4 = 0; s4 < 4; ++s4) acc[s4][r] *= al;
        }
#pragma unroll
        for (int r = 0; r < 8; ++r) {           // row-sum across 16 lanes
            rs[r] += __shfl_xor(rs[r], 1, 32);
            rs[r] += __shfl_xor(rs[r], 2, 32);
            rs[r] += __shfl_xor(rs[r], 4, 32);
            rs[r] += __shfl_xor(rs[r], 8, 32);
            lrow[r] += rs[r];
        }

        // P (C-layout) -> per-wave swizzled LDS tile -> A-fragment gather.
        // Write mapping is the exact inverse of ld_frag_sw4's read mapping:
        //   u16 index = m*32 + ((c ^ swz4(m))<<3) + (col&7), c = col>>3.
        unsigned short* pw = (unsigned short*)(sP + wave * 256);
#pragma unroll
        for (int r = 0; r < 8; ++r) {
            const int m = r + 8 * g;
            const int x = swz4(m);
            pw[m * 32 + ((((n >> 3) ^ x) & 3) << 3) + (n & 7)]       = f2bf(P0[r]);
            pw[m * 32 + ((((2 + (n >> 3)) ^ x) & 3) << 3) + (n & 7)] = f2bf(P1[r]);
        }
        const v16bf Pf = ld_frag_sw4(sP + wave * 256, n, lane);

        // O += P * V  (4 d-subtiles of 16)
#pragma unroll
        for (int s4 = 0; s4 < 4; ++s4) {
            const v16bf Vf = ld_frag_sw4(&sV[buf][0], s4 * 16 + n, lane);
            acc[s4] = wmma_bf16(Pf, Vf, acc[s4]);
        }
        wait_async0();
        __syncthreads();
    }

    // epilogue: normalize and write f32 out[b][q][h*64+d]
#pragma unroll
    for (int s4 = 0; s4 < 4; ++s4) {
#pragma unroll
        for (int r = 0; r < 8; ++r) {
            const int q = blockIdx.x * 64 + wave * 16 + r + 8 * g;
            const int d = s4 * 16 + n;
            out[(b * SEQ + q) * (NH * DH) + h * DH + d] = acc[s4][r] / lrow[r];
        }
    }
}

// ---------------------------------------------------------------------------
// Host side
// ---------------------------------------------------------------------------
extern "C" void kernel_launch(void* const* d_in, const int* in_sizes, int n_in,
                              void* d_out, int out_size, void* d_ws, size_t ws_size,
                              hipStream_t stream) {
    const float* Q    = (const float*)d_in[0];
    const float* K    = (const float*)d_in[1];
    const float* V    = (const float*)d_in[2];
    const int*   mask = (const int*)d_in[3];
    const float* Wq   = (const float*)d_in[4];
    const float* bq   = (const float*)d_in[5];
    const float* Wk   = (const float*)d_in[6];
    const float* Wv   = (const float*)d_in[7];
    const float* bv   = (const float*)d_in[8];
    float* out = (float*)d_out;

    char* ws = (char*)d_ws;
    const size_t szQ = (size_t)BATCH * SEQ * DM * 2;   // 16 MiB bf16
    const size_t szW = (size_t)DM * DM * 2;            // 2 MiB bf16
    unsigned short* Qb  = (unsigned short*)(ws);
    unsigned short* Kb  = (unsigned short*)(ws + szQ);
    unsigned short* Vb  = (unsigned short*)(ws + 2 * szQ);
    unsigned short* Wqb = (unsigned short*)(ws + 3 * szQ);
    unsigned short* Wkb = (unsigned short*)(ws + 3 * szQ + szW);
    unsigned short* Wvb = (unsigned short*)(ws + 3 * szQ + 2 * szW);
    unsigned short* qpj = (unsigned short*)(ws + 3 * szQ + 3 * szW);
    unsigned short* kpj = (unsigned short*)(ws + 4 * szQ + 3 * szW);
    unsigned short* vtj = (unsigned short*)(ws + 5 * szQ + 3 * szW);

    const int nQ = BATCH * SEQ * DM;   // 8388608
    const int nW = DM * DM;            // 1048576

    mha_cvt_bf16<<<nQ / 1024, 256, 0, stream>>>(Q,  Qb);
    mha_cvt_bf16<<<nQ / 1024, 256, 0, stream>>>(K,  Kb);
    mha_cvt_bf16<<<nQ / 1024, 256, 0, stream>>>(V,  Vb);
    mha_cvt_bf16<<<nW / 1024, 256, 0, stream>>>(Wq, Wqb);
    mha_cvt_bf16<<<nW / 1024, 256, 0, stream>>>(Wk, Wkb);
    mha_cvt_bf16<<<nW / 1024, 256, 0, stream>>>(Wv, Wvb);

    const dim3 gGemm(BATCH * SEQ / 256, NH);
    mha_proj_gemm<<<gGemm, 256, 0, stream>>>(Qb, Wqb, bq,      qpj, 0);
    mha_proj_gemm<<<gGemm, 256, 0, stream>>>(Kb, Wkb, nullptr, kpj, 0);
    mha_proj_gemm<<<gGemm, 256, 0, stream>>>(Vb, Wvb, bv,      vtj, 1);

    const dim3 gAttn(SEQ / 64, BATCH * NH);
    mha_attn<<<gAttn, 128, 0, stream>>>(qpj, kpj, vtj, mask, out);
}
